// CNN_Experts_v1_73143293051396
// MI455X (gfx1250) — matrix operand order repaired and checked by reading, not compiled, not run on hardware
//
#include <hip/hip_runtime.h>

typedef _Float16 half16 __attribute__((ext_vector_type(16)));
typedef _Float16 half8  __attribute__((ext_vector_type(8)));
typedef float    float8 __attribute__((ext_vector_type(8)));

#define Bq     32
#define Cc     128
#define Hh     56
#define Wd     56
#define HWc    3136
#define KSELc  2
#define KNc    128
#define Ec     8
#define K1c    1152        // 9 * C, stage-1 GEMM depth (reordered: k = rs*128 + c)
#define CKc    256         // C + KN, stage-2 GEMM depth
#define NTc    64          // spatial tile width
#define TILESc 49          // 3136 / 64
#define NSTEP1 36          // K1c / 32
#define NSTEP2 8           // CKc / 32
#define PK1_ELEMS (Ec * NSTEP1 * 128 * 32)        // 1,179,648 f16
#define PK2_ELEMS (Ec * NSTEP2 * 128 * 32)        //   262,144 f16
#define XH_ELEMS  ((size_t)Bq * Cc * HWc)         // 12,845,056 f16

// ---------------------------------------------------------------------------
// One-time pre-pass: weights f32 -> f16 in A-fragment order, x f32 -> f16.
// ---------------------------------------------------------------------------
__global__ __launch_bounds__(256)
void prepack(const float* __restrict__ wspec,
             const float* __restrict__ wch,
             const float* __restrict__ x,
             _Float16*    __restrict__ pk1,
             _Float16*    __restrict__ pk2,
             _Float16*    __restrict__ xh)
{
    size_t id = (size_t)blockIdx.x * 256 + threadIdx.x;
    if (id < PK1_ELEMS) {
        int i  = (int)id;
        int kl = i & 31;
        int r1 = i >> 5;
        int m  = r1 & 127;
        int r2 = r1 >> 7;
        int st = r2 % NSTEP1;
        int e  = r2 / NSTEP1;
        int rs = st >> 2;
        int c  = ((st & 3) << 5) + kl;
        pk1[i] = (_Float16)wspec[(((size_t)(e * 128 + m) * 128 + c) * 9) + rs];
    } else if (id < PK1_ELEMS + PK2_ELEMS) {
        int i  = (int)(id - PK1_ELEMS);
        int kl = i & 31;
        int r1 = i >> 5;
        int m  = r1 & 127;
        int r2 = r1 >> 7;
        int st = r2 & 7;
        int e  = r2 >> 3;
        pk2[i] = (_Float16)wch[((size_t)(e * 128 + m) * CKc) + (st << 5) + kl];
    } else {
        size_t i = id - PK1_ELEMS - PK2_ELEMS;
        if (i < XH_ELEMS) xh[i] = (_Float16)x[i];
    }
}

// ---------------------------------------------------------------------------
// Fused MoE conv. Stage 1: implicit-GEMM 3x3 conv (M=128, K=1152 reordered),
// Stage 2: 1x1 conv (M=128, K=256) over [x ; nf].
// Hand-pipelined double buffering, compile-time buffer selection.
// ---------------------------------------------------------------------------
__global__ __launch_bounds__(256)
void moe_conv_fused_wmma(const _Float16* __restrict__ xh,
                         const int*      __restrict__ gate,
                         const _Float16* __restrict__ pk1,
                         const _Float16* __restrict__ pk2,
                         float*          __restrict__ out)
{
    __shared__ __align__(32) _Float16 lds_b1[2][4][32][16];  // 8 KB, double buffer
    __shared__ __align__(32) _Float16 lds_cat[8][4][32][16]; // 32 KB

    const int tid   = threadIdx.x;
    const int lane  = tid & 31;
    const int wv    = tid >> 5;          // wave id 0..7 (wave32)
    const int nmod  = lane & 15;
    const int nhalf = lane >> 4;

    const int tile   = blockIdx.x;       // 0..48
    const int bk     = blockIdx.y;       // b*KSEL + kcol
    const int b      = bk >> 1;
    const int hwbase = tile * NTc;

    const int idx = gate[bk];

    const _Float16* xhb = xh + (size_t)b * Cc * HWc;

    // Producer role: thread owns spatial n = tid>>2, K-quarter q = tid&3.
    const int q     = tid & 3;
    const int np    = tid >> 2;               // 0..63
    const int hwoff = hwbase + np;
    const int hp    = hwoff / Wd;
    const int wp    = hwoff - hp * Wd;
    const int b1off = (((np >> 4) * 32 + (((q >> 1) << 4) | (np & 15))) << 4)
                      | ((q & 1) << 3);       // half-index inside one buffer
    _Float16* const b1st  = &lds_b1[0][0][0][0] + b1off;   // store base, buf stride 2048
    const _Float16* const b1ld = &lds_b1[0][0][0][0] + (lane << 4); // load base

    auto produce = [&](int step) -> half8 {
        int rs    = step >> 2;                // 0..8
        int r     = (rs * 11) >> 5;           // rs / 3
        int sc    = rs - r * 3;               // rs % 3
        int cbase = (step & 3) << 5;
        int  hh   = hp + r - 1;
        int  ww   = wp + sc - 1;
        bool inb  = ((unsigned)hh < (unsigned)Hh) & ((unsigned)ww < (unsigned)Wd);
        int  soff = inb ? (hwoff + (r - 1) * Wd + (sc - 1)) : hwoff;
        const _Float16* s = xhb + (size_t)(cbase + (q << 3)) * HWc + soff;
        half8 hv;
        #pragma unroll
        for (int e = 0; e < 8; ++e) {
            _Float16 v = s[(size_t)e * HWc];
            hv[e] = inb ? v : (_Float16)0.f;
        }
        return hv;
    };

    const _Float16* a1p = pk1 + ((size_t)idx * NSTEP1 * 128 + (wv * 16 + nmod)) * 32
                              + nhalf * 16;
    const _Float16* a2p = pk2 + ((size_t)idx * NSTEP2 * 128 + (wv * 16 + nmod)) * 32
                              + nhalf * 16;

    // ---------------- Stage 1 ----------------
    float8 acc1[4] = {};

    // One pipelined step: consume buffer CUR for `step`, optionally produce
    // `step+1` into buffer CUR^1. CUR is a compile-time literal at every call.
    auto s1body = [&](int step, int CUR, bool PROD) {
        const _Float16* ap = a1p + (size_t)step * 4096;
        half8 alo = *reinterpret_cast<const half8*>(ap);
        half8 ahi = *reinterpret_cast<const half8*>(ap + 8);

        half8 hvn;
        if (PROD) hvn = produce(step + 1);

        const _Float16* bb = b1ld + CUR * 2048;
        half16 bf0 = *reinterpret_cast<const half16*>(bb);
        half16 bf1 = *reinterpret_cast<const half16*>(bb + 512);
        half16 bf2 = *reinterpret_cast<const half16*>(bb + 1024);
        half16 bf3 = *reinterpret_cast<const half16*>(bb + 1536);

        half16 a = __builtin_shufflevector(alo, ahi,
                      0,1,2,3,4,5,6,7,8,9,10,11,12,13,14,15);
        acc1[0] = __builtin_amdgcn_wmma_f32_16x16x32_f16(false, a, false, bf0,
                                                         (short)0, acc1[0], false, false);
        acc1[1] = __builtin_amdgcn_wmma_f32_16x16x32_f16(false, a, false, bf1,
                                                         (short)0, acc1[1], false, false);
        acc1[2] = __builtin_amdgcn_wmma_f32_16x16x32_f16(false, a, false, bf2,
                                                         (short)0, acc1[2], false, false);
        acc1[3] = __builtin_amdgcn_wmma_f32_16x16x32_f16(false, a, false, bf3,
                                                         (short)0, acc1[3], false, false);

        if (PROD)
            *reinterpret_cast<half8*>(b1st + (CUR ^ 1) * 2048) = hvn;
        __syncthreads();
    };

    // Prologue: fill buffer 0 with step 0.
    {
        half8 hv0 = produce(0);
        *reinterpret_cast<half8*>(b1st) = hv0;
    }
    __syncthreads();

    // Steady state: steps 0..33, two per iteration, constant buffer indices.
    #pragma unroll 1
    for (int it = 0; it < (NSTEP1 - 2) / 2; ++it) {
        const int step = it * 2;
        s1body(step,     0, true);
        s1body(step + 1, 1, true);
    }
    // Epilogue: step 34 (produce 35), step 35 (consume only).
    s1body(NSTEP1 - 2, 0, true);
    s1body(NSTEP1 - 1, 1, false);

    // nf -> lds_cat rows 128..255: 8 halves contiguous -> one b128 per tile.
    {
        const int ks    = 4 + (wv >> 1);
        const int lidx  = ((wv & 1) << 4) | nmod;
        const int ebase = nhalf << 3;          // 0 or 8
        #pragma unroll
        for (int t = 0; t < 4; ++t) {
            half8 hv;
            #pragma unroll
            for (int i = 0; i < 8; ++i) hv[i] = (_Float16)acc1[t][i];
            *reinterpret_cast<half8*>(&lds_cat[ks][t][lidx][ebase]) = hv;
        }
    }

    // x-tile -> lds_cat rows 0..127 (f16 vector loads, 8 consecutive n each).
    #pragma unroll
    for (int t = 0; t < 4; ++t) {
        int g  = tid + 256 * t;                // 0..1023
        int c  = g >> 3;                       // 0..127
        int n8 = (g & 7) << 3;                 // 0,8,...,56
        half8 v = *reinterpret_cast<const half8*>(xhb + (size_t)c * HWc + hwbase + n8);
        const int ks = c >> 5;
        const int nt = n8 >> 4;
        const int lo = (((c >> 4) & 1) << 4);
        const int e  = c & 15;
        #pragma unroll
        for (int u = 0; u < 8; ++u)
            lds_cat[ks][nt][lo | ((n8 + u) & 15)][e] = v[u];
    }
    __syncthreads();

    // ---------------- Stage 2: fully unrolled, immediate LDS offsets --------
    float8 acc2[4] = {};
    const _Float16* const catld = &lds_cat[0][0][0][0] + (lane << 4);

    #pragma unroll
    for (int step = 0; step < NSTEP2; ++step) {
        half8 alo = *reinterpret_cast<const half8*>(a2p + (size_t)step * 4096);
        half8 ahi = *reinterpret_cast<const half8*>(a2p + (size_t)step * 4096 + 8);

        const _Float16* bb = catld + step * 2048;
        half16 bf0 = *reinterpret_cast<const half16*>(bb);
        half16 bf1 = *reinterpret_cast<const half16*>(bb + 512);
        half16 bf2 = *reinterpret_cast<const half16*>(bb + 1024);
        half16 bf3 = *reinterpret_cast<const half16*>(bb + 1536);

        half16 a = __builtin_shufflevector(alo, ahi,
                      0,1,2,3,4,5,6,7,8,9,10,11,12,13,14,15);
        acc2[0] = __builtin_amdgcn_wmma_f32_16x16x32_f16(false, a, false, bf0,
                                                         (short)0, acc2[0], false, false);
        acc2[1] = __builtin_amdgcn_wmma_f32_16x16x32_f16(false, a, false, bf1,
                                                         (short)0, acc2[1], false, false);
        acc2[2] = __builtin_amdgcn_wmma_f32_16x16x32_f16(false, a, false, bf2,
                                                         (short)0, acc2[2], false, false);
        acc2[3] = __builtin_amdgcn_wmma_f32_16x16x32_f16(false, a, false, bf3,
                                                         (short)0, acc2[3], false, false);
    }

    // Epilogue: f32 stores, coalesced in n within each 16-lane half.
    float* ob = out + (size_t)bk * Cc * HWc;
    #pragma unroll
    for (int t = 0; t < 4; ++t) {
        #pragma unroll
        for (int i = 0; i < 8; ++i) {
            int m  = (lane < 16) ? i : (8 + i);
            int o2 = wv * 16 + m;
            int n  = t * 16 + nmod;
            ob[(size_t)o2 * HWc + hwbase + n] = acc2[t][i];
        }
    }
}

extern "C" void kernel_launch(void* const* d_in, const int* in_sizes, int n_in,
                              void* d_out, int out_size, void* d_ws, size_t ws_size,
                              hipStream_t stream) {
    (void)in_sizes; (void)n_in; (void)out_size; (void)ws_size;
    const float* x     = (const float*)d_in[0];
    const int*   gate  = (const int*)  d_in[1];
    const float* wspec = (const float*)d_in[2];
    const float* wch   = (const float*)d_in[3];
    float*       out   = (float*)d_out;

    _Float16* pk1 = (_Float16*)d_ws;                 // 2.25 MB
    _Float16* pk2 = pk1 + PK1_ELEMS;                 // 0.5 MB
    _Float16* xh  = pk2 + PK2_ELEMS;                 // 25.7 MB (ws >= ~28.5 MB)

    size_t ntot = (size_t)PK1_ELEMS + PK2_ELEMS + XH_ELEMS;
    prepack<<<(int)((ntot + 255) / 256), 256, 0, stream>>>(wspec, wch, x, pk1, pk2, xh);

    dim3 grid(TILESc, Bq * KSELc);   // 49 spatial tiles x 64 (b, gate-col) pairs
    moe_conv_fused_wmma<<<grid, 256, 0, stream>>>(xh, gate, pk1, pk2, out);
}
